// EcaSparseAttention_59090160058555
// MI455X (gfx1250) — matrix operand order repaired and checked by reading, not compiled, not run on hardware
//
#include <hip/hip_runtime.h>
#include <cmath>

// ---------------------------------------------------------------------------
// EcaSparseAttention for MI455X (gfx1250, wave32).
// Memory-bound: streams W1 (393MB) + W2 (118MB) once -> ~22us floor @23.3TB/s.
// GEMMs: activations pre-packed to zero-padded f16 (16 x Kpad); the WMMA hot
// loop is branchless (2x b128 A + 4x b128 W + cvt + wmma), unrolled 4x so the
// scheduler can overlap loads of later tiles with earlier WMMAs. The single
// partial K-tail tile (GEMM2 last chunk only) is hoisted out of the loop.
// ---------------------------------------------------------------------------

#define BATCH   8
#define CCH     30
#define LPS     16384      // H*H
#define LIMITN  8192       // top-k count
#define NH1     6000
#define NH2     4914
#define NBINS   4096
#define KPAD1   16384      // GEMM1 padded K (multiple of 32)
#define KPAD2   6016       // GEMM2 padded K (4*47*32)

typedef __attribute__((ext_vector_type(16))) _Float16 v16h;
typedef __attribute__((ext_vector_type(8)))  float    v8f;

union FRAG16 { v16h v; _Float16 e[16]; uint4 q[2]; };

// ---- Stage 1: adaptive avg pool per (b, c) --------------------------------
__global__ __launch_bounds__(256)
void pool_kernel(const float* __restrict__ adj, float* __restrict__ pooled) {
    __shared__ float red[256];
    const int bc = blockIdx.x;                       // 0..239
    const float* base = adj + (long)bc * LPS;
    float s = 0.0f;
    for (int i = threadIdx.x; i < LPS; i += 256) s += base[i];
    red[threadIdx.x] = s;
    __syncthreads();
    for (int off = 128; off > 0; off >>= 1) {
        if (threadIdx.x < off) red[threadIdx.x] += red[threadIdx.x + off];
        __syncthreads();
    }
    if (threadIdx.x == 0) pooled[bc] = red[0] * (1.0f / (float)LPS);
}

// ---- Stage 2: ECA conv(k=3) + sigmoid -> scale (also band_score output) ---
__global__ __launch_bounds__(256)
void eca_kernel(const float* __restrict__ pooled, const float* __restrict__ ecaw,
                float* __restrict__ scale, float* __restrict__ band_out) {
    int tid = threadIdx.x;
    if (tid >= BATCH * CCH) return;
    int b = tid / CCH, c = tid % CCH;
    float s = 0.0f;
    #pragma unroll
    for (int k = 0; k < 3; ++k) {
        int cc = c + k - 1;
        if (cc >= 0 && cc < CCH) s += pooled[b * CCH + cc] * ecaw[k];
    }
    float sg = 1.0f / (1.0f + expf(-s));
    scale[tid]    = sg;
    band_out[tid] = sg;          // second tuple output
}

// ---- Stage 3: y = sum_c adj*scale, plus blockwise min/max partials --------
__global__ __launch_bounds__(256)
void wsum_kernel(const float* __restrict__ adj, const float* __restrict__ scale,
                 float* __restrict__ y, float* __restrict__ pmn, float* __restrict__ pmx) {
    __shared__ float sc[CCH];
    __shared__ float rmn[256], rmx[256];
    const int b = blockIdx.y;
    const int i = blockIdx.x * 256 + threadIdx.x;    // < LPS
    if (threadIdx.x < CCH) sc[threadIdx.x] = scale[b * CCH + threadIdx.x];
    __syncthreads();
    float s = 0.0f;
    #pragma unroll 5
    for (int c = 0; c < CCH; ++c)
        s += adj[((long)(b * CCH + c)) * LPS + i] * sc[c];
    y[b * LPS + i] = s;
    rmn[threadIdx.x] = s; rmx[threadIdx.x] = s;
    __syncthreads();
    for (int off = 128; off > 0; off >>= 1) {
        if (threadIdx.x < off) {
            rmn[threadIdx.x] = fminf(rmn[threadIdx.x], rmn[threadIdx.x + off]);
            rmx[threadIdx.x] = fmaxf(rmx[threadIdx.x], rmx[threadIdx.x + off]);
        }
        __syncthreads();
    }
    if (threadIdx.x == 0) { pmn[b * 64 + blockIdx.x] = rmn[0]; pmx[b * 64 + blockIdx.x] = rmx[0]; }
}

__global__ __launch_bounds__(64)
void minmax_kernel(const float* __restrict__ pmn, const float* __restrict__ pmx,
                   float* __restrict__ mnmx) {
    __shared__ float smn[64], smx[64];
    const int b = blockIdx.x, t = threadIdx.x;
    smn[t] = pmn[b * 64 + t]; smx[t] = pmx[b * 64 + t];
    __syncthreads();
    for (int off = 32; off > 0; off >>= 1) {
        if (t < off) { smn[t] = fminf(smn[t], smn[t + off]); smx[t] = fmaxf(smx[t], smx[t + off]); }
        __syncthreads();
    }
    if (t == 0) { mnmx[b * 2] = smn[0]; mnmx[b * 2 + 1] = smx[0]; }
}

// ---- Stage 4: v = norm(y); score = sigmoid(conv13(v)); histogram ----------
__global__ __launch_bounds__(256)
void zero_i32_kernel(int* __restrict__ p, int n) {
    int i = blockIdx.x * 256 + threadIdx.x;
    if (i < n) p[i] = 0;
}

__global__ __launch_bounds__(256)
void score_kernel(const float* __restrict__ y, const float* __restrict__ mnmx,
                  const float* __restrict__ attw, float* __restrict__ v,
                  float* __restrict__ score, int* __restrict__ hist) {
    const int b = blockIdx.y;
    const int i = blockIdx.x * 256 + threadIdx.x;
    const float mn = mnmx[b * 2];
    const float inv = 1.0f / (mnmx[b * 2 + 1] - mn);
    const float* yb = y + b * LPS;
    const float vi = (yb[i] - mn) * inv;
    v[b * LPS + i] = vi;
    float s = 0.0f;
    #pragma unroll
    for (int k = 0; k < 13; ++k) {
        int j = i + k - 6;
        if (j >= 0 && j < LPS) s += (yb[j] - mn) * inv * attw[k];   // zero-padded conv
    }
    float sg = 1.0f / (1.0f + expf(-s));
    score[b * LPS + i] = sg;
    int bin = (int)(sg * (float)NBINS);
    bin = bin < 0 ? 0 : (bin > NBINS - 1 ? NBINS - 1 : bin);
    atomicAdd(&hist[b * NBINS + bin], 1);
}

// ---- Stage 5: find threshold bin + quota (top-LIMITN selection) -----------
__global__ void thresh_kernel(const int* __restrict__ hist, int* __restrict__ tinfo) {
    if (threadIdx.x != 0) return;
    const int b = blockIdx.x;
    int cum = 0;
    for (int bin = NBINS - 1; bin >= 0; --bin) {
        int h = hist[b * NBINS + bin];
        if (cum + h >= LIMITN) {
            tinfo[b * 4 + 0] = bin;          // threshold bin
            tinfo[b * 4 + 1] = LIMITN - cum; // quota inside threshold bin
            tinfo[b * 4 + 2] = 0;            // quota counter
            return;
        }
        cum += h;
    }
}

// ---- Stage 6: scatter kept values: sparse = v*(score+1) on top-k ----------
__global__ __launch_bounds__(256)
void sparse_kernel(const float* __restrict__ v, const float* __restrict__ score,
                   int* __restrict__ tinfo, float* __restrict__ sparse) {
    const int b = blockIdx.y;
    const int idx = b * LPS + blockIdx.x * 256 + threadIdx.x;
    const float sg = score[idx];
    int bin = (int)(sg * (float)NBINS);
    bin = bin < 0 ? 0 : (bin > NBINS - 1 ? NBINS - 1 : bin);
    const int tbin = tinfo[b * 4 + 0];
    float outv = 0.0f;
    if (bin > tbin) {
        outv = v[idx] * (sg + 1.0f);
    } else if (bin == tbin) {
        int pos = atomicAdd(&tinfo[b * 4 + 2], 1);
        if (pos < tinfo[b * 4 + 1]) outv = v[idx] * (sg + 1.0f);
    }
    sparse[idx] = outv;
}

// ---- A-repack: fp32 (BATCH x K) -> f16 (16 x Kpad), rows/K zero-padded ----
__global__ __launch_bounds__(256)
void pack_a16_kernel(const float* __restrict__ A, _Float16* __restrict__ A16,
                     int K, int Kpad) {
    const int i = blockIdx.x * 256 + threadIdx.x;    // over 16*Kpad
    if (i >= 16 * Kpad) return;
    const int m = i / Kpad, k = i % Kpad;
    const float x = (m < BATCH && k < K) ? A[m * K + k] : 0.0f;
    A16[i] = (_Float16)x;
}

// ---- Bias init: Out[b][n] = bias[n] (partials accumulate on top) ----------
__global__ __launch_bounds__(256)
void bias_init_kernel(const float* __restrict__ bias, float* __restrict__ out, int N, int total) {
    int i = blockIdx.x * 256 + threadIdx.x;
    if (i < total) out[i] = bias[i % N];
}

// ---- helpers for the GEMM fragments ---------------------------------------
__device__ __forceinline__ void cvt16(FRAG16& bf, const float4 w0, const float4 w1,
                                      const float4 w2, const float4 w3) {
    bf.e[0]  = (_Float16)w0.x; bf.e[1]  = (_Float16)w0.y;
    bf.e[2]  = (_Float16)w0.z; bf.e[3]  = (_Float16)w0.w;
    bf.e[4]  = (_Float16)w1.x; bf.e[5]  = (_Float16)w1.y;
    bf.e[6]  = (_Float16)w1.z; bf.e[7]  = (_Float16)w1.w;
    bf.e[8]  = (_Float16)w2.x; bf.e[9]  = (_Float16)w2.y;
    bf.e[10] = (_Float16)w2.z; bf.e[11] = (_Float16)w2.w;
    bf.e[12] = (_Float16)w3.x; bf.e[13] = (_Float16)w3.y;
    bf.e[14] = (_Float16)w3.z; bf.e[15] = (_Float16)w3.w;
}

// ---- Stage 7/8: skinny GEMM  Out[m][n] += sum_k A[m][k] * W[n][k] ----------
// One wave per (16-wide N tile, K chunk). Branchless hot loop, unrolled 4x.
// A-frag: contiguous k order, same as B (lane=M; lanes0-15 K kt+0..15, lanes16-31 K kt+16..31).
// B-frag: same contiguous k order as A (lane=N; lanes0-15 K kt+0..15, lanes16-31 K kt+16..31).
__global__ __launch_bounds__(32)
void gemm_atb_wmma(const _Float16* __restrict__ A16, const float* __restrict__ W,
                   float* __restrict__ Out, int K, int Kpad, int Nout, int ktiles) {
    const int lane = threadIdx.x;                     // 0..31
    const int n = blockIdx.x * 16 + (lane & 15);
    const bool nvalid = (n < Nout);
    const int nclamp = nvalid ? n : (Nout - 1);       // safe row, store masked
    const int alow = (lane >> 4) ? 8  : 0;
    const int blow = (lane >> 4) ? 16 : 0;
    const float*    wrow = W + (long)nclamp * K;
    const _Float16* arow = A16 + (lane & 15) * Kpad;  // rows 8..15 are zeros
    const int k0 = blockIdx.y * ktiles * 32;

    // Wave-uniform tile accounting: 'full' branchless tiles + at most one tail
    int rem = K - k0;
    if (rem < 0) rem = 0;
    int full = rem >> 5;
    if (full > ktiles) full = ktiles;
    const int tailk = (full < ktiles) ? (rem - (full << 5)) : 0;   // 0..31

    v8f acc = {};
    #pragma unroll 4
    for (int t = 0; t < full; ++t) {
        const int kt = k0 + t * 32;
        FRAG16 af, bf;
        af.q[0] = *(const uint4*)(arow + kt + blow);          // 8 f16, 16B aligned
        af.q[1] = *(const uint4*)(arow + kt + blow + 8);
        const float4* wp = (const float4*)(wrow + kt + blow); // 16 f32, 16B aligned
        const float4 w0 = wp[0], w1 = wp[1], w2 = wp[2], w3 = wp[3];
        cvt16(bf, w0, w1, w2, w3);
        acc = __builtin_amdgcn_wmma_f32_16x16x32_f16(
            false, af.v, false, bf.v, (short)0, acc, false, false);
    }

    if (tailk > 0) {                                  // wave-uniform; runs once
        const int kt = k0 + full * 32;
        FRAG16 af, bf;
        af.q[0] = *(const uint4*)(arow + kt + blow);  // A16 zero-padded to Kpad
        af.q[1] = *(const uint4*)(arow + kt + blow + 8);
        #pragma unroll
        for (int j = 0; j < 16; ++j) {
            const int kb = blow + j;
            bf.e[j] = (_Float16)((kb < tailk) ? wrow[kt + kb] : 0.0f);
        }
        acc = __builtin_amdgcn_wmma_f32_16x16x32_f16(
            false, af.v, false, bf.v, (short)0, acc, false, false);
    }

    // D layout: lanes 0..15 hold M = vgpr index (0..7) -> exactly our 8 rows.
    if (lane < 16 && nvalid) {
        #pragma unroll
        for (int j = 0; j < BATCH; ++j)
            atomicAdd(&Out[j * Nout + n], acc[j]);
    }
}

// ---- Stage 9: classifier head + log_softmax -------------------------------
__global__ __launch_bounds__(256)
void head_kernel(const float* __restrict__ h2, const float* __restrict__ Wc1,
                 const float* __restrict__ Wc2, float* __restrict__ out) {
    __shared__ float t[BATCH][32];
    __shared__ float logits[BATCH][5];
    const int tid = threadIdx.x;                    // 256 = 8 * 32
    const int b = tid >> 5, j = tid & 31;
    const float* hrow = h2 + b * NH2;
    const float* wrow = Wc1 + j * NH2;
    float s = 0.0f;
    for (int k = 0; k < NH2; ++k) s += hrow[k] * wrow[k];
    t[b][j] = s;
    __syncthreads();
    if (tid < BATCH * 5) {
        const int bb = tid / 5, c = tid % 5;
        float acc = 0.0f;
        #pragma unroll
        for (int jj = 0; jj < 32; ++jj) acc += t[bb][jj] * Wc2[c * 32 + jj];
        logits[bb][c] = acc;
    }
    __syncthreads();
    if (tid < BATCH) {
        float mx = -1e30f;
        #pragma unroll
        for (int c = 0; c < 5; ++c) mx = fmaxf(mx, logits[tid][c]);
        float se = 0.0f;
        #pragma unroll
        for (int c = 0; c < 5; ++c) se += expf(logits[tid][c] - mx);
        const float lse = logf(se) + mx;
        #pragma unroll
        for (int c = 0; c < 5; ++c) out[tid * 5 + c] = logits[tid][c] - lse;
    }
}

// ---------------------------------------------------------------------------
extern "C" void kernel_launch(void* const* d_in, const int* in_sizes, int n_in,
                              void* d_out, int out_size, void* d_ws, size_t ws_size,
                              hipStream_t stream) {
    const float* adj   = (const float*)d_in[0];
    const float* eca_w = (const float*)d_in[1];
    const float* att_w = (const float*)d_in[2];
    const float* W1    = (const float*)d_in[3];
    const float* b1    = (const float*)d_in[4];
    const float* W2    = (const float*)d_in[5];
    const float* b2    = (const float*)d_in[6];
    const float* Wc1   = (const float*)d_in[7];
    const float* Wc2   = (const float*)d_in[8];

    float* out_log  = (float*)d_out;           // (8,5) log-softmax
    float* out_band = out_log + BATCH * 5;     // (8,30) band_score

    // Workspace layout (floats, 16B-aligned chunks); total ~3.3 MB.
    float* F      = (float*)d_ws;
    float* pooled = F;                          // 240 (pad 256)
    float* scale  = F + 256;                    // 240 (pad 256)
    float* yv     = F + 512;                    // B*LPS = 131072
    float* pmn    = yv + BATCH * LPS;           // 512
    float* pmx    = pmn + 512;                  // 512
    float* mnmx   = pmx + 512;                  // 16
    float* vv     = mnmx + 16;                  // 131072
    float* score  = vv + BATCH * LPS;           // 131072
    float* sparse = score + BATCH * LPS;        // 131072
    float* h1     = sparse + BATCH * LPS;       // 8*6000 = 48000
    float* h2     = h1 + BATCH * NH1;           // 8*4914 = 39312
    int*   hist   = (int*)(h2 + BATCH * NH2);   // 8*4096
    int*   tinfo  = hist + BATCH * NBINS;       // 8*4 (pad 32)
    _Float16* a16_1 = (_Float16*)(tinfo + 32);  // 16*KPAD1 halves (512KB)
    _Float16* a16_2 = a16_1 + 16 * KPAD1;       // 16*KPAD2 halves (188KB)

    // 1) channel pool + ECA attention (band_score)
    pool_kernel<<<BATCH * CCH, 256, 0, stream>>>(adj, pooled);
    eca_kernel<<<1, 256, 0, stream>>>(pooled, eca_w, scale, out_band);

    // 2) weighted channel sum + min/max normalize
    wsum_kernel<<<dim3(LPS / 256, BATCH), 256, 0, stream>>>(adj, scale, yv, pmn, pmx);
    minmax_kernel<<<BATCH, 64, 0, stream>>>(pmn, pmx, mnmx);

    // 3) attention score + histogram top-k sparsification
    zero_i32_kernel<<<(BATCH * NBINS + 255) / 256, 256, 0, stream>>>(hist, BATCH * NBINS);
    score_kernel<<<dim3(LPS / 256, BATCH), 256, 0, stream>>>(yv, mnmx, att_w, vv, score, hist);
    thresh_kernel<<<BATCH, 32, 0, stream>>>(hist, tinfo);
    sparse_kernel<<<dim3(LPS / 256, BATCH), 256, 0, stream>>>(vv, score, tinfo, sparse);

    // 4) h1 = sparse @ W1.T + b1  (streams 393MB of W1 -> bandwidth bound)
    pack_a16_kernel<<<(16 * KPAD1 + 255) / 256, 256, 0, stream>>>(sparse, a16_1, LPS, KPAD1);
    bias_init_kernel<<<(BATCH * NH1 + 255) / 256, 256, 0, stream>>>(b1, h1, NH1, BATCH * NH1);
    gemm_atb_wmma<<<dim3(NH1 / 16, 8), 32, 0, stream>>>(a16_1, W1, h1, LPS, KPAD1, NH1, 64);

    // 5) h2 = h1 @ W2.T + b2  (streams 118MB of W2)
    pack_a16_kernel<<<(16 * KPAD2 + 255) / 256, 256, 0, stream>>>(h1, a16_2, NH1, KPAD2);
    bias_init_kernel<<<(BATCH * NH2 + 255) / 256, 256, 0, stream>>>(b2, h2, NH2, BATCH * NH2);
    gemm_atb_wmma<<<dim3((NH2 + 15) / 16, 4), 32, 0, stream>>>(a16_2, W2, h2, NH1, KPAD2, NH2, 47);

    // 6) classifier head + log-softmax
    head_kernel<<<1, 256, 0, stream>>>(h2, Wc1, Wc2, out_log);

    (void)in_sizes; (void)n_in; (void)out_size; (void)ws_size;
}
